// THR_layer_68487548502056
// MI455X (gfx1250) — compile-verified, matrix operand order checked by reading, not run on hardware
//
#include <hip/hip_runtime.h>
#include <hip/hip_bf16.h>

typedef _Float16 half_t;
typedef __attribute__((ext_vector_type(16))) _Float16 v16h;
typedef __attribute__((ext_vector_type(8)))  _Float16 v8h;
typedef __attribute__((ext_vector_type(8)))  float    v8f;

#define EPSV   1e-5f
#define CCH    512
#define NHEADC 8
#define DHEAD  64
#define HDIM   2048

enum { MODE_F16 = 0, MODE_GELU = 1, MODE_PROJ = 2, MODE_FINAL = 3, MODE_F16T = 4 };

// CDNA5 wave32 16-bit fragment: halves [0..7] cover K = h8..h8+7,
// halves [8..15] cover K = 16+h8..16+h8+7  (h8 = 8 for lanes 16..31, else 0).
// => one fragment == two contiguous 16-byte chunks when K is the fast axis.
__device__ __forceinline__ v8h ld8(const half_t* p) { return *(const v8h*)p; }
__device__ __forceinline__ v16h mkfrag(v8h lo, v8h hi) {
    return __builtin_shufflevector(lo, hi, 0,1,2,3,4,5,6,7,8,9,10,11,12,13,14,15);
}

// ---------------------------------------------------------------------------
__global__ __launch_bounds__(256)
void f2h_k(const float* __restrict__ src, half_t* __restrict__ dst, int n) {
    int i = blockIdx.x * 256 + threadIdx.x;
    if (i < n) dst[i] = (half_t)src[i];
}

// ---------------------------------------------------------------------------
// Channel LayerNorm of raw input (B,C,T) -> out0 in token-major (N, C) f32
__global__ __launch_bounds__(128)
void cln_input_k(const float* __restrict__ x, const float* __restrict__ lw,
                 const float* __restrict__ lb, float* __restrict__ out0,
                 int T, int N)
{
    __shared__ float red[128];
    const int n = blockIdx.x;
    const int b = n / T, t = n - b * T;
    const int tid = threadIdx.x;
    float vals[4];
    float s = 0.f;
#pragma unroll
    for (int i = 0; i < 4; ++i) {
        const int c = tid + i * 128;
        vals[i] = x[(b * CCH + c) * T + t];
        s += vals[i];
    }
    red[tid] = s; __syncthreads();
    for (int o = 64; o > 0; o >>= 1) { if (tid < o) red[tid] += red[tid + o]; __syncthreads(); }
    const float mu = red[0] * (1.f / CCH);
    __syncthreads();
    float s2 = 0.f;
#pragma unroll
    for (int i = 0; i < 4; ++i) { float r = vals[i] - mu; s2 += r * r; }
    red[tid] = s2; __syncthreads();
    for (int o = 64; o > 0; o >>= 1) { if (tid < o) red[tid] += red[tid + o]; __syncthreads(); }
    const float rs = rsqrtf(red[0] * (1.f / CCH) + EPSV);
#pragma unroll
    for (int i = 0; i < 4; ++i) {
        const int c = tid + i * 128;
        out0[n * CCH + c] = (vals[i] - mu) * rs * lw[c] + lb[c];
    }
}

// ---------------------------------------------------------------------------
// phi stage 1: partial sums over t-slices, coalesced over channels
__global__ __launch_bounds__(512)
void phi1_k(const float* __restrict__ out0, float* __restrict__ part, int T)
{
    const int s = blockIdx.x;            // 16 slices
    const int b = blockIdx.y;
    const int c = threadIdx.x;
    const int tps = T / 16;
    const int t0 = s * tps;
    float acc = 0.f;
    for (int t = 0; t < tps; ++t)
        acc += out0[(b * T + t0 + t) * CCH + c];
    part[(b * 16 + s) * CCH + c] = acc;
}
// phi stage 2: phi[b,c] = relu(mean * gfc_w + gfc_b)
__global__ __launch_bounds__(512)
void phi2_k(const float* __restrict__ part, const float* __restrict__ gw,
            const float* __restrict__ gb, float* __restrict__ phi, int T)
{
    const int b = blockIdx.x;
    const int c = threadIdx.x;
    float s = 0.f;
#pragma unroll
    for (int i = 0; i < 16; ++i) s += part[(b * 16 + i) * CCH + c];
    const float v = (s / (float)T) * gw[c] + gb[c];
    phi[b * CCH + c] = v > 0.f ? v : 0.f;
}

// ---------------------------------------------------------------------------
// Depthwise convs fused:  xatt = (cw+ckw)*psi ;  ykv = (out*fc_w+fc_b)*phi
// out0/xatt/ykv in (N, C); channel-fast thread mapping (coalesced).
__global__ __launch_bounds__(256)
void mix_k(const float* __restrict__ out0, const float* __restrict__ phi,
           const float* __restrict__ psi_w, const float* __restrict__ psi_b,
           const float* __restrict__ fc_w, const float* __restrict__ fc_b,
           const float* __restrict__ cw_w, const float* __restrict__ cw_b,
           const float* __restrict__ ckw_w, const float* __restrict__ ckw_b, int kk,
           half_t* __restrict__ xatt, half_t* __restrict__ ykv, int T, int N)
{
    const int idx = blockIdx.x * 256 + threadIdx.x;
    if (idx >= CCH * N) return;
    const int c = idx & (CCH - 1), n = idx >> 9;
    const int b = n / T, t = n - b * T;
    const int base = n * CCH + c;
    const float f0  = out0[base];
    const float fm1 = (t - 1 >= 0) ? out0[base - CCH] : 0.f;
    const float fp1 = (t + 1 <  T) ? out0[base + CCH] : 0.f;
    float psi = psi_b[c] + psi_w[c*3+0]*fm1 + psi_w[c*3+1]*f0 + psi_w[c*3+2]*fp1;
    float cw  = cw_b[c]  + cw_w[c*3+0]*fm1  + cw_w[c*3+1]*f0  + cw_w[c*3+2]*fp1;
    float ckw = ckw_b[c];
    const int h = kk >> 1;
    for (int j = 0; j < kk; ++j) {
        const int tt = t + j - h;
        const float fv = (tt >= 0 && tt < T) ? out0[base + (j - h) * CCH] : 0.f;
        ckw += ckw_w[c*kk+j] * fv;
    }
    const float fcv = f0 * fc_w[c] + fc_b[c];
    xatt[base] = (half_t)((cw + ckw) * psi);
    ykv[base]  = (half_t)(fcv * phi[b * CCH + c]);
}

// ---------------------------------------------------------------------------
// q/k/v prep: depthwise conv3 (no bias) * mask, then channel LN. (N,C) layout.
__global__ __launch_bounds__(128)
void qkv_prep_k(const half_t* __restrict__ src, const float* __restrict__ dw,
                const float* __restrict__ nw, const float* __restrict__ nb,
                const unsigned char* __restrict__ mask,
                half_t* __restrict__ dst, int T, int N)
{
    __shared__ float red[128];
    const int n = blockIdx.x;
    const int b = n / T, t = n - b * T;
    const int tid = threadIdx.x;
    const float mk = mask[n] ? 1.f : 0.f;
    float vals[4];
    float s = 0.f;
#pragma unroll
    for (int i = 0; i < 4; ++i) {
        const int c = tid + i * 128;
        const int base = n * CCH + c;
        const float vm1 = (t > 0)     ? (float)src[base - CCH] : 0.f;
        const float v0  = (float)src[base];
        const float vp1 = (t < T - 1) ? (float)src[base + CCH] : 0.f;
        vals[i] = (dw[c*3+0]*vm1 + dw[c*3+1]*v0 + dw[c*3+2]*vp1) * mk;
        s += vals[i];
    }
    red[tid] = s; __syncthreads();
    for (int o = 64; o > 0; o >>= 1) { if (tid < o) red[tid] += red[tid + o]; __syncthreads(); }
    const float mu = red[0] * (1.f / CCH);
    __syncthreads();
    float s2 = 0.f;
#pragma unroll
    for (int i = 0; i < 4; ++i) { float r = vals[i] - mu; s2 += r * r; }
    red[tid] = s2; __syncthreads();
    for (int o = 64; o > 0; o >>= 1) { if (tid < o) red[tid] += red[tid + o]; __syncthreads(); }
    const float rs = rsqrtf(red[0] * (1.f / CCH) + EPSV);
#pragma unroll
    for (int i = 0; i < 4; ++i) {
        const int c = tid + i * 128;
        dst[n * CCH + c] = (half_t)((vals[i] - mu) * rs * nw[c] + nb[c]);
    }
}

// ---------------------------------------------------------------------------
// GroupNorm(16) over (32 ch, T); out2 (N,C) f32 -> gno (N,C) f16
__global__ __launch_bounds__(256)
void gn_k(const float* __restrict__ x, const float* __restrict__ gw,
          const float* __restrict__ gb, half_t* __restrict__ y, int T, int N)
{
    __shared__ float rs1[256], rs2[256];
    const int g = blockIdx.x & 15;
    const int b = blockIdx.x >> 4;
    const int c0 = g * 32;
    const int tid = threadIdx.x;
    const int tot = 32 * T;
    float s = 0.f, s2 = 0.f;
    for (int i = tid; i < tot; i += 256) {
        const int c = c0 + (i & 31), t = i >> 5;
        const float v = x[(b * T + t) * CCH + c];
        s += v; s2 += v * v;
    }
    rs1[tid] = s; rs2[tid] = s2; __syncthreads();
    for (int o = 128; o > 0; o >>= 1) {
        if (tid < o) { rs1[tid] += rs1[tid + o]; rs2[tid] += rs2[tid + o]; }
        __syncthreads();
    }
    const float mu  = rs1[0] / (float)tot;
    const float var = rs2[0] / (float)tot - mu * mu;
    const float rsq = rsqrtf(var + EPSV);
    for (int i = tid; i < tot; i += 256) {
        const int c = c0 + (i & 31), t = i >> 5;
        const float v = x[(b * T + t) * CCH + c];
        y[(b * T + t) * CCH + c] = (half_t)((v - mu) * rsq * gw[c] + gb[c]);
    }
}

// ---------------------------------------------------------------------------
// WMMA GEMM: Y[m,n] = sum_k W[m,k]*X[n,k] + bias[m]; W row-major (M,K),
// X token-major (N,K) -> all fragments are pairs of b128 loads.
// 4 waves/block, each wave a 32x32 tile (2x2 WMMA tiles).
__global__ __launch_bounds__(128)
void gemm_wmma_k(const half_t* __restrict__ W, const half_t* __restrict__ X,
                 const float* __restrict__ bias,
                 int M, int K, int N, int T, int mode,
                 half_t* __restrict__ Y16, float* __restrict__ Yf,
                 const float* __restrict__ R0, const float* __restrict__ xres,
                 const unsigned char* __restrict__ mask)
{
    const int lane = threadIdx.x & 31;
    const int wave = threadIdx.x >> 5;
    const int row0 = blockIdx.y * 64 + (wave >> 1) * 32;
    const int col0 = blockIdx.x * 64 + (wave & 1) * 32;
    const int nloc = lane & 15;
    const int h8 = (lane & 16) ? 8 : 0;

    const half_t* wp0 = W + (row0 + nloc) * K + h8;
    const half_t* wp1 = W + (row0 + 16 + nloc) * K + h8;
    const half_t* xp0 = X + (col0 + nloc) * K + h8;
    const half_t* xp1 = X + (col0 + 16 + nloc) * K + h8;

    v8f acc[2][2];
#pragma unroll
    for (int i = 0; i < 2; ++i)
#pragma unroll
        for (int j = 0; j < 2; ++j)
#pragma unroll
            for (int r = 0; r < 8; ++r) acc[i][j][r] = 0.f;

    for (int k0 = 0; k0 < K; k0 += 32) {
        const v16h a0 = mkfrag(ld8(wp0 + k0), ld8(wp0 + k0 + 16));
        const v16h a1 = mkfrag(ld8(wp1 + k0), ld8(wp1 + k0 + 16));
        const v16h b0 = mkfrag(ld8(xp0 + k0), ld8(xp0 + k0 + 16));
        const v16h b1 = mkfrag(ld8(xp1 + k0), ld8(xp1 + k0 + 16));
        acc[0][0] = __builtin_amdgcn_wmma_f32_16x16x32_f16(false, a0, false, b0, (short)0, acc[0][0], false, false);
        acc[0][1] = __builtin_amdgcn_wmma_f32_16x16x32_f16(false, a0, false, b1, (short)0, acc[0][1], false, false);
        acc[1][0] = __builtin_amdgcn_wmma_f32_16x16x32_f16(false, a1, false, b0, (short)0, acc[1][0], false, false);
        acc[1][1] = __builtin_amdgcn_wmma_f32_16x16x32_f16(false, a1, false, b1, (short)0, acc[1][1], false, false);
    }

#pragma unroll
    for (int i = 0; i < 2; ++i)
#pragma unroll
        for (int j = 0; j < 2; ++j) {
            const int mb = row0 + i * 16 + h8;          // 8 consecutive rows mb..mb+7
            const int n  = col0 + j * 16 + nloc;
            if (mode == MODE_F16 || mode == MODE_GELU) {
                v8h o;
#pragma unroll
                for (int r = 0; r < 8; ++r) {
                    float val = acc[i][j][r] + bias[mb + r];
                    if (mode == MODE_GELU)
                        val = 0.5f * val * (1.f + erff(val * 0.70710678118f));
                    o[r] = (half_t)val;
                }
                *(v8h*)(Y16 + n * M + mb) = o;          // (N,M) layout, one b128
            } else if (mode == MODE_F16T) {             // (M,N) layout (for V)
#pragma unroll
                for (int r = 0; r < 8; ++r)
                    Y16[(mb + r) * N + n] = (half_t)(acc[i][j][r] + bias[mb + r]);
            } else if (mode == MODE_PROJ) {
                const int b = n / T, t = n - b * T;
                const float mkv = mask[n] ? 1.f : 0.f;
#pragma unroll
                for (int r = 0; r < 8; ++r) {
                    const float val = acc[i][j][r] + bias[mb + r];
                    Yf[n * M + mb + r] = val * mkv + R0[n * M + mb + r]
                                       + xres[(b * M + mb + r) * T + t] * mkv;
                }
            } else {                                    // MODE_FINAL -> (B,C,T)
                const int b = n / T, t = n - b * T;
#pragma unroll
                for (int r = 0; r < 8; ++r)
                    Yf[(b * M + mb + r) * T + t] =
                        acc[i][j][r] + bias[mb + r] + R0[n * M + mb + r];
            }
        }
}

// ---------------------------------------------------------------------------
// Flash attention: one wave per 16 query rows per (head, batch).
// q1/k1 token-major (N,C); v1 channel-major (C,N); atto token-major (N,C).
__global__ __launch_bounds__(32)
void attn_k(const half_t* __restrict__ q1, const half_t* __restrict__ k1,
            const half_t* __restrict__ v1, const unsigned char* __restrict__ mask,
            half_t* __restrict__ att_out, int T, int N)
{
    __shared__ half_t pbuf[16 * 32];
    const int lane = threadIdx.x;
    const int nloc = lane & 15;
    const int h8 = (lane & 16) ? 8 : 0;
    const int tq0 = blockIdx.x * 16;
    const int hb = blockIdx.y * DHEAD;
    const int b = blockIdx.z;
    const int bT = b * T;
    const float scale = 0.125f;  // 1/sqrt(64)

    // Q fragments (K = head-dim, contiguous in (N,C)): 2 b128 loads each
    v16h qa[2];
#pragma unroll
    for (int ks = 0; ks < 2; ++ks) {
        const half_t* qp = q1 + (bT + tq0 + nloc) * CCH + hb + ks * 32 + h8;
        qa[ks] = mkfrag(ld8(qp), ld8(qp + 16));
#pragma unroll
        for (int e = 0; e < 16; ++e) qa[ks][e] = qa[ks][e] * (half_t)0.125f;
        (void)scale;
    }

    v8f acc[4];
#pragma unroll
    for (int j = 0; j < 4; ++j)
#pragma unroll
        for (int r = 0; r < 8; ++r) acc[j][r] = 0.f;

    float m_r[8], l_r[8];
#pragma unroll
    for (int r = 0; r < 8; ++r) { m_r[r] = -1e30f; l_r[r] = 0.f; }

    for (int s0 = 0; s0 < T; s0 += 32) {
        v8f S0, S1;
#pragma unroll
        for (int r = 0; r < 8; ++r) { S0[r] = 0.f; S1[r] = 0.f; }
#pragma unroll
        for (int ks = 0; ks < 2; ++ks) {
            const half_t* kp0 = k1 + (bT + s0 + nloc) * CCH + hb + ks * 32 + h8;
            const half_t* kp1 = kp0 + 16 * CCH;
            const v16h kb0 = mkfrag(ld8(kp0), ld8(kp0 + 16));
            const v16h kb1 = mkfrag(ld8(kp1), ld8(kp1 + 16));
            S0 = __builtin_amdgcn_wmma_f32_16x16x32_f16(false, qa[ks], false, kb0, (short)0, S0, false, false);
            S1 = __builtin_amdgcn_wmma_f32_16x16x32_f16(false, qa[ks], false, kb1, (short)0, S1, false, false);
        }
        const float am0 = mask[bT + s0 + nloc] ? 0.f : -1e30f;
        const float am1 = mask[bT + s0 + 16 + nloc] ? 0.f : -1e30f;
#pragma unroll
        for (int r = 0; r < 8; ++r) {
            const float a0v = S0[r] + am0;
            const float a1v = S1[r] + am1;
            float mx = fmaxf(a0v, a1v);
#pragma unroll
            for (int off = 8; off >= 1; off >>= 1) mx = fmaxf(mx, __shfl_xor(mx, off, 32));
            const float nm = fmaxf(m_r[r], mx);
            const float p0 = __expf(a0v - nm);
            const float p1 = __expf(a1v - nm);
            float rsum = p0 + p1;
#pragma unroll
            for (int off = 8; off >= 1; off >>= 1) rsum += __shfl_xor(rsum, off, 32);
            const float al = __expf(m_r[r] - nm);
            l_r[r] = l_r[r] * al + rsum;
            m_r[r] = nm;
#pragma unroll
            for (int j = 0; j < 4; ++j) acc[j][r] *= al;
            const int row = r + h8;
            pbuf[row * 32 + nloc]      = (half_t)p0;
            pbuf[row * 32 + 16 + nloc] = (half_t)p1;
        }
        __syncthreads();
        const half_t* pp = pbuf + nloc * 32 + h8;
        const v16h pa = mkfrag(ld8(pp), ld8(pp + 16));
        __syncthreads();
#pragma unroll
        for (int j = 0; j < 4; ++j) {
            const half_t* vp = v1 + (hb + j * 16 + nloc) * N + bT + s0 + h8;
            v16h vb = mkfrag(ld8(vp), ld8(vp + 16));
#pragma unroll
            for (int e = 0; e < 16; ++e) {
                const int k = (e < 8) ? (h8 + e) : (16 + h8 + e - 8);
                if (!mask[bT + s0 + k]) vb[e] = (half_t)0.f;
            }
            acc[j] = __builtin_amdgcn_wmma_f32_16x16x32_f16(false, pa, false, vb, (short)0, acc[j], false, false);
        }
    }
#pragma unroll
    for (int r = 0; r < 8; ++r) {
        const float inv = l_r[r] > 0.f ? 1.f / l_r[r] : 0.f;
#pragma unroll
        for (int j = 0; j < 4; ++j)
            att_out[(bT + tq0 + r + h8) * CCH + hb + j * 16 + nloc] =
                (half_t)(acc[j][r] * inv);
    }
}

// ---------------------------------------------------------------------------
extern "C" void kernel_launch(void* const* d_in, const int* in_sizes, int n_in,
                              void* d_out, int out_size, void* d_ws, size_t ws_size,
                              hipStream_t stream)
{
    (void)in_sizes; (void)n_in; (void)out_size; (void)ws_size;
    const float* x4 = (const float*)d_in[0];
    const float* x8 = (const float*)d_in[1];
    const float* xh = (const float*)d_in[2];
    const unsigned char* m4 = (const unsigned char*)d_in[3];
    const unsigned char* m8 = (const unsigned char*)d_in[4];
    const unsigned char* mh = (const unsigned char*)d_in[5];
    const float* ln_w = (const float*)d_in[6];
    const float* ln_b = (const float*)d_in[7];
    const float* gn_w = (const float*)d_in[8];
    const float* gn_b = (const float*)d_in[9];
    const float* psi_w = (const float*)d_in[10];
    const float* psi_b = (const float*)d_in[11];
    const float* fc_w = (const float*)d_in[12];
    const float* fc_b = (const float*)d_in[13];
    const float* convw_w = (const float*)d_in[14];
    const float* convw_b = (const float*)d_in[15];
    const float* ckwL_w = (const float*)d_in[16];
    const float* ckwL_b = (const float*)d_in[17];
    const float* ckwM_w = (const float*)d_in[18];
    const float* ckwM_b = (const float*)d_in[19];
    const float* ckwH_w = (const float*)d_in[20];
    const float* ckwH_b = (const float*)d_in[21];
    const float* gfc_w = (const float*)d_in[22];
    const float* gfc_b = (const float*)d_in[23];
    const float* ca_dw_w = (const float*)d_in[24];
    const float* ca_n_w = (const float*)d_in[25];
    const float* ca_n_b = (const float*)d_in[26];
    const float* ca_qkv_w = (const float*)d_in[27];
    const float* ca_qkv_b = (const float*)d_in[28];
    const float* ca_proj_w = (const float*)d_in[29];
    const float* ca_proj_b = (const float*)d_in[30];
    const float* mlp1_w = (const float*)d_in[31];
    const float* mlp1_b = (const float*)d_in[32];
    const float* mlp2_w = (const float*)d_in[33];
    const float* mlp2_b = (const float*)d_in[34];

    const int B = 2, C = CCH;
    const int Ts[3] = {512, 1024, 2048};
    const int Nmax = B * 2048;

    char* wsp = (char*)d_ws;
    size_t off = 0;
    auto alloc = [&](size_t bytes) -> void* {
        void* p = wsp + off;
        off += (bytes + 255) & ~(size_t)255;
        return p;
    };
    half_t* qkvW16  = (half_t*)alloc((size_t)3 * C * C * 2);
    half_t* projW16 = (half_t*)alloc((size_t)C * C * 2);
    half_t* mlp1W16 = (half_t*)alloc((size_t)HDIM * C * 2);
    half_t* mlp2W16 = (half_t*)alloc((size_t)C * HDIM * 2);
    float*  out0 = (float*)alloc((size_t)C * Nmax * 4);   // (N,C)
    float*  phi  = (float*)alloc((size_t)B * C * 4);
    float*  part = (float*)alloc((size_t)B * 16 * C * 4);
    half_t* xatt = (half_t*)alloc((size_t)C * Nmax * 2);  // (N,C)
    half_t* ykv  = (half_t*)alloc((size_t)C * Nmax * 2);  // (N,C)
    half_t* q0   = (half_t*)alloc((size_t)C * Nmax * 2);  // (N,C)
    half_t* k0   = (half_t*)alloc((size_t)C * Nmax * 2);
    half_t* v0   = (half_t*)alloc((size_t)C * Nmax * 2);
    half_t* q1   = (half_t*)alloc((size_t)C * Nmax * 2);  // (N,C)
    half_t* k1   = (half_t*)alloc((size_t)C * Nmax * 2);  // (N,C)
    half_t* v1   = (half_t*)alloc((size_t)C * Nmax * 2);  // (C,N)
    half_t* atto = (half_t*)alloc((size_t)C * Nmax * 2);  // (N,C)
    float*  out2 = (float*)alloc((size_t)C * Nmax * 4);   // (N,C)
    half_t* gno  = (half_t*)alloc((size_t)C * Nmax * 2);  // (N,C)
    half_t* h1   = (half_t*)alloc((size_t)HDIM * Nmax * 2); // (N,HDIM)

    f2h_k<<<(3 * C * C + 255) / 256, 256, 0, stream>>>(ca_qkv_w, qkvW16, 3 * C * C);
    f2h_k<<<(C * C + 255) / 256, 256, 0, stream>>>(ca_proj_w, projW16, C * C);
    f2h_k<<<(HDIM * C + 255) / 256, 256, 0, stream>>>(mlp1_w, mlp1W16, HDIM * C);
    f2h_k<<<(C * HDIM + 255) / 256, 256, 0, stream>>>(mlp2_w, mlp2W16, C * HDIM);

    const float* xs_arr[3] = {x4, x8, xh};
    const unsigned char* m_arr[3] = {m4, m8, mh};
    const float* ckw_w_arr[3] = {ckwL_w, ckwM_w, ckwH_w};
    const float* ckw_b_arr[3] = {ckwL_b, ckwM_b, ckwH_b};
    const int kk_arr[3] = {1, 3, 5};
    size_t out_off = 0;

    for (int br = 0; br < 3; ++br) {
        const int T = Ts[br];
        const int N = B * T;
        const float* xin = xs_arr[br];
        const unsigned char* mk = m_arr[br];

        cln_input_k<<<N, 128, 0, stream>>>(xin, ln_w, ln_b, out0, T, N);
        phi1_k<<<dim3(16, B), 512, 0, stream>>>(out0, part, T);
        phi2_k<<<B, 512, 0, stream>>>(part, gfc_w, gfc_b, phi, T);
        mix_k<<<(C * N + 255) / 256, 256, 0, stream>>>(
            out0, phi, psi_w, psi_b, fc_w, fc_b,
            convw_w + br * C * 3, convw_b + br * C,
            ckw_w_arr[br], ckw_b_arr[br], kk_arr[br], xatt, ykv, T, N);

        qkv_prep_k<<<N, 128, 0, stream>>>(xatt, ca_dw_w + 0 * C * 3, ca_n_w + 0 * C, ca_n_b + 0 * C, mk, q0, T, N);
        qkv_prep_k<<<N, 128, 0, stream>>>(ykv,  ca_dw_w + 1 * C * 3, ca_n_w + 1 * C, ca_n_b + 1 * C, mk, k0, T, N);
        qkv_prep_k<<<N, 128, 0, stream>>>(ykv,  ca_dw_w + 2 * C * 3, ca_n_w + 2 * C, ca_n_b + 2 * C, mk, v0, T, N);

        const dim3 blk(128);
        gemm_wmma_k<<<dim3(N / 64, C / 64), blk, 0, stream>>>(
            qkvW16 + 0 * C * C, q0, ca_qkv_b + 0 * C, C, C, N, T, MODE_F16,
            q1, nullptr, nullptr, nullptr, nullptr);
        gemm_wmma_k<<<dim3(N / 64, C / 64), blk, 0, stream>>>(
            qkvW16 + 1 * C * C, k0, ca_qkv_b + 1 * C, C, C, N, T, MODE_F16,
            k1, nullptr, nullptr, nullptr, nullptr);
        gemm_wmma_k<<<dim3(N / 64, C / 64), blk, 0, stream>>>(
            qkvW16 + 2 * C * C, v0, ca_qkv_b + 2 * C, C, C, N, T, MODE_F16T,
            v1, nullptr, nullptr, nullptr, nullptr);

        attn_k<<<dim3(T / 16, NHEADC, B), 32, 0, stream>>>(q1, k1, v1, mk, atto, T, N);

        gemm_wmma_k<<<dim3(N / 64, C / 64), blk, 0, stream>>>(
            projW16, atto, ca_proj_b, C, C, N, T, MODE_PROJ,
            nullptr, out2, out0, xin, mk);

        gn_k<<<B * 16, 256, 0, stream>>>(out2, gn_w, gn_b, gno, T, N);

        gemm_wmma_k<<<dim3(N / 64, HDIM / 64), blk, 0, stream>>>(
            mlp1W16, gno, mlp1_b, HDIM, C, N, T, MODE_GELU,
            h1, nullptr, nullptr, nullptr, nullptr);

        float* dOut = (float*)d_out + out_off;
        gemm_wmma_k<<<dim3(N / 64, C / 64), blk, 0, stream>>>(
            mlp2W16, h1, mlp2_b, C, HDIM, N, T, MODE_FINAL,
            nullptr, dOut, out2, nullptr, nullptr);

        out_off += (size_t)B * C * T;
    }
}
